// RNNModel_46600395162112
// MI455X (gfx1250) — matrix operand order repaired and checked by reading, not compiled, not run on hardware
//
#include <hip/hip_runtime.h>
#include <hip/hip_bf16.h>

#define T_    64
#define B_    32
#define NHID  768
#define NB    6
#define BS    128
#define TOPK  4
#define NTOK  32000
#define ATT   512
#define DK    64
#define NH2   4
#define DK2   16
#define G4    512   // 4*BS

typedef float  v2f   __attribute__((ext_vector_type(2)));
typedef float  v4f   __attribute__((ext_vector_type(4)));
typedef float  v8f   __attribute__((ext_vector_type(8)));
typedef __bf16 v8bf  __attribute__((ext_vector_type(8)));
typedef __bf16 v16bf __attribute__((ext_vector_type(16)));

// ---------------------------------------------------------------------------
// Exact f32 GEMM tile via V_WMMA_F32_16X16X4_F32 (direct-global operands).
// A layout: lane r=row, lanes 0-15 K={k,k+1}, lanes 16-31 K={k+2,k+3}.
// B layout: lane r=col, VGPR j -> K=k+j (lo half) / K=k+2+j (hi half).
// C/D: VGPR j = row j+8*half, col = lane%16.
// ---------------------------------------------------------------------------
__device__ __forceinline__ v8f wmma_f32_gemm(const float* __restrict__ A, int lda,
                                             const float* __restrict__ B, int ldb,
                                             int K, v8f acc, float ascale) {
  const int lane = threadIdx.x & 31;
  const int r    = lane & 15;
  const int h2   = (lane >> 4) << 1;   // 0 or 2
  const float* arow = A + (size_t)r * lda;
  for (int k = 0; k < K; k += 4) {
    v2f a, b;
    a.x = arow[k + h2 + 0] * ascale;
    a.y = arow[k + h2 + 1] * ascale;
    b.x = B[(size_t)(k + h2 + 0) * ldb + r];
    b.y = B[(size_t)(k + h2 + 1) * ldb + r];
    acc = __builtin_amdgcn_wmma_f32_16x16x4_f32(false, a, false, b,
                                                (short)0, acc, false, false);
  }
  return acc;
}

// Async 16B global->LDS copy (ASYNCcnt-tracked CDNA5 VGLOBAL async path).
// ISA (15.18 op 98): GLOBAL_LOAD_ASYNC_TO_LDS_B128 vdst(LDS byte addr),
// vaddr(64b global), saddr=off. Flat-LDS aperture rule: LDS byte address is
// the low 32 bits of the flat pointer.
__device__ __forceinline__ void async_b128(const float* g, float* l) {
  unsigned lds = (unsigned)(uintptr_t)l;
  asm volatile("global_load_async_to_lds_b128 %0, %1, off"
               :: "v"(lds), "v"(g) : "memory");
}

// ---- per-step kernel 1: k1 = xt @ Wk_in[1], v1 = xt @ Wv_in[1] -------------
__global__ void k_kv(const float* __restrict__ x, int t,
                     const float* __restrict__ Wk1, const float* __restrict__ Wv1,
                     float* __restrict__ k1, float* __restrict__ v1) {
  const float* xt = x + (size_t)t * B_ * NHID;
  int wid  = blockIdx.x * (blockDim.x >> 5) + (threadIdx.x >> 5);   // 0..71
  int lane = threadIdx.x & 31;
  int r = lane & 15, half = lane >> 4;
  const float* Bp; int ldb; float* out; int ldc; int m0, n0;
  if (wid < 8) { int mt = wid >> 2, nt = wid & 3;  m0 = mt*16; n0 = nt*16;
                 Bp = Wk1 + n0; ldb = DK;  out = k1; ldc = DK; }
  else         { int v = wid - 8; int mt = v >> 5, nt = v & 31; m0 = mt*16; n0 = nt*16;
                 Bp = Wv1 + n0; ldb = ATT; out = v1; ldc = ATT; }
  v8f acc = {};
  acc = wmma_f32_gemm(xt + (size_t)m0 * NHID, NHID, Bp, ldb, NHID, acc, 1.0f);
  for (int j = 0; j < 8; ++j)
    out[(size_t)(m0 + j + 8*half) * ldc + n0 + r] = acc[j];
}

// ---- per-step kernel 2: scores, p1=sigmoid(s), top-k drop mask -------------
__global__ void k_score_mask(const float* __restrict__ hx,
                             const float* __restrict__ Wq_in,
                             const float* __restrict__ k1,
                             float* __restrict__ p1, float* __restrict__ mask) {
  __shared__ float snull[B_][NB];
  int tid = threadIdx.x;
  if (tid < B_ * NB) {
    int b = tid / NB, n = tid % NB;
    const float* hb = hx + b * NHID + n * BS;
    const float* W  = Wq_in + (size_t)n * BS * DK;
    const float* kb = k1 + b * DK;
    float s = 0.f;
    for (int d = 0; d < BS; ++d) {
      float acc = 0.f;
      for (int kk = 0; kk < DK; ++kk) acc += W[d * DK + kk] * kb[kk];
      s += hb[d] * acc;
    }
    s *= 0.125f;                       // 1/sqrt(DK)
    float pg = 1.f / (1.f + expf(-s)); // softmax([0,s])[1]
    p1[b * NB + n]  = pg;
    snull[b][n]     = 1.f - pg;
  }
  __syncthreads();
  if (tid < B_) {
    float nl[NB], mk[NB];
    for (int n = 0; n < NB; ++n) { nl[n] = snull[tid][n]; mk[n] = 1.f; }
    for (int d = 0; d < NB - TOPK; ++d) {   // drop 2 largest null (tie: low idx)
      int best = 0; float bv = -1e30f;
      for (int n = 0; n < NB; ++n)
        if (mk[n] > 0.5f && nl[n] > bv) { bv = nl[n]; best = n; }
      mk[best] = 0.f;
    }
    for (int n = 0; n < NB; ++n) mask[tid * NB + n] = mk[n];
  }
}

// ---- per-step kernel 3: gates = p1*(v1@Wih[n]) + hb@Whh[n] + b_lstm --------
// WG = (n, mt): 16 rows x 512 cols. Wih/Whh K-chunks (8x512 f32 = 16KB,
// contiguous) streamed to LDS with async global->LDS b128, double-buffered
// on ASYNCcnt. Wave w owns 4 N-tiles; A fragment loaded once per K-step and
// reused across the 4 tiles. LDS B reads: col%64 unique per lane -> no bank
// conflicts.
__global__ void k_gates(const float* __restrict__ v1, const float* __restrict__ hx,
                        const float* __restrict__ p1,
                        const float* __restrict__ Wih, const float* __restrict__ Whh,
                        const float* __restrict__ b_lstm, float* __restrict__ gates) {
  __shared__ __align__(16) float Bl[2][8][G4];   // 32 KB double buffer
  const int NCH0 = ATT / 8;                      // 64 chunks (Wih part)
  const int NCH  = NCH0 + BS / 8;                // + 16 chunks (Whh part) = 80
  const int tid  = threadIdx.x;
  const int n    = blockIdx.x >> 1;
  const int m0   = (blockIdx.x & 1) * 16;
  const int w    = tid >> 5;
  const int lane = tid & 31;
  const int r    = lane & 15, half = lane >> 4, h2 = half << 1;
  const float ps = p1[(m0 + r) * NB + n];

  const float* Wih_n = Wih + (size_t)n * ATT * G4;
  const float* Whh_n = Whh + (size_t)n * BS * G4;

  // prologue: chunk 0
  for (int j = 0; j < 4; ++j)
    async_b128(Wih_n + (size_t)(tid + j * 256) * 4, &Bl[0][0][0] + (tid + j * 256) * 4);

  v8f acc[4] = {};
  for (int c = 0; c < NCH; ++c) {
    if (c + 1 < NCH) {                            // issue next chunk
      const float* gsrc = (c + 1 < NCH0)
          ? Wih_n + (size_t)(c + 1) * 8 * G4
          : Whh_n + (size_t)(c + 1 - NCH0) * 8 * G4;
      float* lbuf = &Bl[(c + 1) & 1][0][0];
      for (int j = 0; j < 4; ++j)
        async_b128(gsrc + (size_t)(tid + j * 256) * 4, lbuf + (tid + j * 256) * 4);
      asm volatile("s_wait_asynccnt 0x4" ::: "memory");  // oldest chunk landed
    } else {
      asm volatile("s_wait_asynccnt 0x0" ::: "memory");
    }
    __syncthreads();

    const bool p0   = (c < NCH0);
    const float* Ab = p0 ? (v1 + (size_t)m0 * ATT) : (hx + (size_t)m0 * NHID + n * BS);
    const int lda   = p0 ? ATT : NHID;
    const int kb    = p0 ? c * 8 : (c - NCH0) * 8;
    const float asc = p0 ? ps : 1.0f;
    const float* arow = Ab + (size_t)r * lda + kb;
    const float(*Bc)[G4] = Bl[c & 1];
    for (int kk = 0; kk < 8; kk += 4) {
      v2f a;
      a.x = arow[kk + h2 + 0] * asc;
      a.y = arow[kk + h2 + 1] * asc;
#pragma unroll
      for (int tt = 0; tt < 4; ++tt) {
        int col = w * 64 + tt * 16 + r;
        v2f b;
        b.x = Bc[kk + h2 + 0][col];
        b.y = Bc[kk + h2 + 1][col];
        acc[tt] = __builtin_amdgcn_wmma_f32_16x16x4_f32(false, a, false, b,
                                                        (short)0, acc[tt],
                                                        false, false);
      }
    }
    __syncthreads();
  }

  const float* bl = b_lstm + n * G4;
#pragma unroll
  for (int tt = 0; tt < 4; ++tt) {
    int n0 = w * 64 + tt * 16;
    for (int j = 0; j < 8; ++j) {
      int b = m0 + j + 8 * half;
      gates[((size_t)(b * NB + n)) * G4 + n0 + r] = acc[tt][j] + bl[n0 + r];
    }
  }
}

// ---- per-step kernel 4: LSTM nonlinearity ----------------------------------
__global__ void k_lstm(const float* __restrict__ gates, const float* __restrict__ cx,
                       float* __restrict__ cnew, float* __restrict__ hpre) {
  int idx = blockIdx.x * blockDim.x + threadIdx.x;
  if (idx >= B_ * NB * BS) return;
  int h  = idx & (BS - 1);
  int bn = idx >> 7;                      // b*NB + n
  int b = bn / NB, n = bn % NB;
  const float* g = gates + (size_t)bn * G4;
  float gi = g[h], gf = g[BS + h], gg = g[2*BS + h], go = g[3*BS + h];
  float cb = cx[b * NHID + n * BS + h];
  float si = 1.f/(1.f+expf(-gi)), sf = 1.f/(1.f+expf(-gf)), so = 1.f/(1.f+expf(-go));
  float cn = sf * cb + si * tanhf(gg);
  cnew[idx] = cn;
  hpre[idx] = so * tanhf(cn);
}

// ---- per-step kernel 5: q2/k2/v2 = h_pre @ W{q,k,v}_c[n] -------------------
__global__ void k_qkv2(const float* __restrict__ hpre,
                       const float* __restrict__ Wq_c, const float* __restrict__ Wk_c,
                       const float* __restrict__ Wv_c,
                       float* __restrict__ q2, float* __restrict__ k2,
                       float* __restrict__ v2) {
  int wid  = blockIdx.x * (blockDim.x >> 5) + (threadIdx.x >> 5);   // 0..143
  int lane = threadIdx.x & 31;
  int r = lane & 15, half = lane >> 4;
  int w = wid / 48, rem = wid % 48;
  int n = rem / 8, rem2 = rem % 8;
  int m0 = (rem2 >> 2) * 16, n0 = (rem2 & 3) * 16;
  const float* W = (w == 0 ? Wq_c : w == 1 ? Wk_c : Wv_c) + (size_t)n * BS * 64;
  float* out     = (w == 0 ? q2   : w == 1 ? k2   : v2);
  v8f acc = {};
  acc = wmma_f32_gemm(hpre + (size_t)m0 * (NB*BS) + n * BS, NB*BS,
                      W + n0, 64, BS, acc, 1.0f);
  for (int j = 0; j < 8; ++j) {
    int b = m0 + j + 8*half;
    out[((size_t)(b * NB + n)) * 64 + n0 + r] = acc[j];
  }
}

// ---- per-step kernel 6: 6x6 block-comm attention per (batch, head) ---------
__global__ void k_comm(const float* __restrict__ q2, const float* __restrict__ k2,
                       const float* __restrict__ v2, float* __restrict__ comm) {
  int tid = threadIdx.x;
  if (tid >= B_ * NH2) return;
  int b = tid >> 2, h = tid & 3;
  float sc[NB][NB];
  for (int i = 0; i < NB; ++i)
    for (int j = 0; j < NB; ++j) {
      const float* qv = q2 + ((size_t)(b*NB + i)) * 64 + h * DK2;
      const float* kv = k2 + ((size_t)(b*NB + j)) * 64 + h * DK2;
      float s = 0.f;
      for (int k = 0; k < DK2; ++k) s += qv[k] * kv[k];
      sc[i][j] = s * 0.25f;            // 1/sqrt(DK2)
    }
  for (int i = 0; i < NB; ++i) {
    float mx = sc[i][0];
    for (int j = 1; j < NB; ++j) mx = fmaxf(mx, sc[i][j]);
    float den = 0.f;
    for (int j = 0; j < NB; ++j) { sc[i][j] = expf(sc[i][j] - mx); den += sc[i][j]; }
    float inv = 1.f / den;
    for (int k = 0; k < DK2; ++k) {
      float a = 0.f;
      for (int j = 0; j < NB; ++j)
        a += sc[i][j] * v2[((size_t)(b*NB + j)) * 64 + h * DK2 + k];
      comm[((size_t)(b*NB + i)) * 64 + h * DK2 + k] = a * inv;
    }
  }
}

// ---- per-step kernel 7: h_new = h_pre + comm@Wo_c[n]; masked state update --
__global__ void k_update(const float* __restrict__ comm, const float* __restrict__ Wo_c,
                         const float* __restrict__ hpre, const float* __restrict__ cnew,
                         const float* __restrict__ mask,
                         float* __restrict__ hx, float* __restrict__ cx,
                         float* __restrict__ outs, int t) {
  int wid  = blockIdx.x * (blockDim.x >> 5) + (threadIdx.x >> 5);   // 0..95
  int lane = threadIdx.x & 31;
  int r = lane & 15, half = lane >> 4;
  int n = wid / 16, rem = wid % 16;
  int m0 = (rem >> 3) * 16, n0 = (rem & 7) * 16;
  v8f acc = {};
  acc = wmma_f32_gemm(comm + (size_t)m0 * (NB*64) + n * 64, NB*64,
                      Wo_c + (size_t)n * 64 * BS + n0, BS, 64, acc, 1.0f);
  float* od = outs + (size_t)t * B_ * NHID;
  for (int j = 0; j < 8; ++j) {
    int b   = m0 + j + 8*half;
    int col = n0 + r;
    int ei  = (b * NB + n) * BS + col;
    int xi  = b * NHID + n * BS + col;
    float hn = hpre[ei] + acc[j];
    float m  = mask[b * NB + n];
    float hv = m * hn        + (1.f - m) * hx[xi];
    float cv = m * cnew[ei]  + (1.f - m) * cx[xi];
    hx[xi] = hv; cx[xi] = cv; od[xi] = hv;
  }
}

// ---- final decode: (T*B,768) @ Wd^T(768,32000) + bd, bf16 WMMA -------------
// A strip staged in LDS as bf16; per-lane A fragments are two 16B-aligned
// 16B runs -> 2x ds b128. B (Wd row) is 64B contiguous per lane -> 4x float4
// + packed cvt. Wd streamed ahead with global_prefetch.
__global__ void k_decode(const float* __restrict__ outs, const float* __restrict__ Wd,
                         const float* __restrict__ bd, float* __restrict__ dec) {
  __shared__ __align__(16) __bf16 Abf[16][NHID + 8];   // 1552B row = 97*16
  const int m0    = blockIdx.y * 16;
  const int nbase = blockIdx.x * 128;
  const int tid   = threadIdx.x;
  for (int i = tid; i < 16 * NHID; i += blockDim.x) {
    int rr = i / NHID, kk = i % NHID;
    Abf[rr][kk] = (__bf16)outs[(size_t)(m0 + rr) * NHID + kk];
  }
  __syncthreads();
  const int lane = tid & 31;
  const int r = lane & 15, half = lane >> 4;
  const int n0 = nbase + (tid >> 5) * 16;
  const float* wrow = Wd + (size_t)(n0 + r) * NHID;
  v8f acc = {};
  for (int k0 = 0; k0 < NHID; k0 += 32) {
    __builtin_prefetch(wrow + k0 + 128, 0, 1);
    // A: two aligned 8-element bf16 runs per lane
    v8bf lo = *(const v8bf*)&Abf[r][k0 + (half ? 16 : 0)];
    v8bf hi = *(const v8bf*)&Abf[r][k0 + (half ? 24 : 8)];
    v16bf a = __builtin_shufflevector(lo, hi, 0,1,2,3,4,5,6,7,
                                              8,9,10,11,12,13,14,15);
    // B: 16 consecutive f32 from Wd row -> bf16
    const v4f* wp = (const v4f*)(wrow + k0 + 16 * half);
    v16bf b;
#pragma unroll
    for (int j = 0; j < 4; ++j) {
      v4f tv = wp[j];
      b[4*j+0] = (__bf16)tv.x; b[4*j+1] = (__bf16)tv.y;
      b[4*j+2] = (__bf16)tv.z; b[4*j+3] = (__bf16)tv.w;
    }
    acc = __builtin_amdgcn_wmma_f32_16x16x32_bf16(false, a, false, b,
                                                  (short)0, acc, false, false);
  }
  float bias = bd[n0 + r];
  for (int j = 0; j < 8; ++j)
    dec[(size_t)(m0 + j + 8*half) * NTOK + n0 + r] = acc[j] + bias;
}

// ---------------------------------------------------------------------------
extern "C" void kernel_launch(void* const* d_in, const int* in_sizes, int n_in,
                              void* d_out, int out_size, void* d_ws, size_t ws_size,
                              hipStream_t stream) {
  const float* x      = (const float*)d_in[0];
  const float* hx0    = (const float*)d_in[1];
  const float* cx0    = (const float*)d_in[2];
  const float* Wq_in  = (const float*)d_in[3];
  const float* Wk_in  = (const float*)d_in[4];
  const float* Wv_in  = (const float*)d_in[5];
  const float* Wih    = (const float*)d_in[6];
  const float* Whh    = (const float*)d_in[7];
  const float* b_lstm = (const float*)d_in[8];
  const float* Wq_c   = (const float*)d_in[9];
  const float* Wk_c   = (const float*)d_in[10];
  const float* Wv_c   = (const float*)d_in[11];
  const float* Wo_c   = (const float*)d_in[12];
  const float* Wd     = (const float*)d_in[13];
  const float* bd     = (const float*)d_in[14];

  float* ws    = (float*)d_ws;
  float* hx    = ws;                               // B*NHID
  float* cx    = hx   + B_ * NHID;                 // B*NHID
  float* k1    = cx   + B_ * NHID;                 // B*DK
  float* v1    = k1   + B_ * DK;                   // B*ATT
  float* p1    = v1   + B_ * ATT;                  // B*NB
  float* mask  = p1   + B_ * NB;                   // B*NB
  float* gates = mask + B_ * NB;                   // B*NB*G4
  float* cnew  = gates + (size_t)B_ * NB * G4;     // B*NB*BS
  float* hpre  = cnew + B_ * NB * BS;              // B*NB*BS
  float* q2    = hpre + B_ * NB * BS;              // B*NB*64
  float* k2    = q2   + B_ * NB * 64;
  float* v2    = k2   + B_ * NB * 64;
  float* comm  = v2   + B_ * NB * 64;
  float* outs  = comm + B_ * NB * 64;              // T*B*NHID

  (void)hipMemcpyAsync(hx, hx0, B_ * NHID * sizeof(float), hipMemcpyDeviceToDevice, stream);
  (void)hipMemcpyAsync(cx, cx0, B_ * NHID * sizeof(float), hipMemcpyDeviceToDevice, stream);

  const float* Wk1 = Wk_in + (size_t)NHID * DK;    // Wk_in[1]
  const float* Wv1 = Wv_in + (size_t)NHID * ATT;   // Wv_in[1]

  for (int t = 0; t < T_; ++t) {
    k_kv        <<<dim3(9),  256, 0, stream>>>(x, t, Wk1, Wv1, k1, v1);
    k_score_mask<<<dim3(1),  256, 0, stream>>>(hx, Wq_in, k1, p1, mask);
    k_gates     <<<dim3(12), 256, 0, stream>>>(v1, hx, p1, Wih, Whh, b_lstm, gates);
    k_lstm      <<<dim3(96), 256, 0, stream>>>(gates, cx, cnew, hpre);
    k_qkv2      <<<dim3(18), 256, 0, stream>>>(hpre, Wq_c, Wk_c, Wv_c, q2, k2, v2);
    k_comm      <<<dim3(1),  128, 0, stream>>>(q2, k2, v2, comm);
    k_update    <<<dim3(12), 256, 0, stream>>>(comm, Wo_c, hpre, cnew, mask, hx, cx, outs, t);
  }
  k_decode<<<dim3(NTOK / 128, (T_ * B_) / 16), 256, 0, stream>>>(outs, Wd, bd, (float*)d_out);

  float* outp = (float*)d_out;
  (void)hipMemcpyAsync(outp + (size_t)T_ * B_ * NTOK,             hx, B_ * NHID * sizeof(float),
                       hipMemcpyDeviceToDevice, stream);
  (void)hipMemcpyAsync(outp + (size_t)T_ * B_ * NTOK + B_ * NHID, cx, B_ * NHID * sizeof(float),
                       hipMemcpyDeviceToDevice, stream);
}
